// CodebookVQ_86294482911904
// MI455X (gfx1250) — matrix operand order repaired and checked by reading, not compiled, not run on hardware
//
#include <hip/hip_runtime.h>

typedef __attribute__((ext_vector_type(16))) _Float16 v16h;
typedef __attribute__((ext_vector_type(8)))  float    v8f;
typedef __attribute__((ext_vector_type(4)))  float    fvec4;

#define VQ_N 524288
#define VQ_K 256
#define VQ_D 64
#define VQ_WAVES (VQ_N / 16)          // 32768 waves, one 16-row tile each

// d_ws layout:
//   [0,      32768)  : embT  f16 [256][64]  (code-major f16 codebook)
//   [32768,  33792)  : e_sq  f32 [256]
//   [33792, 164864)  : per-wave loss partials f32 [32768]

// ---------------------------------------------------------------------------
// Prep: convert codebook to f16 code-major + squared norms.
// ---------------------------------------------------------------------------
__global__ __launch_bounds__(256) void vq_prep(const float* __restrict__ emb,
                                               _Float16* __restrict__ embT,
                                               float* __restrict__ esq) {
    const int k = threadIdx.x;            // code 0..255
    float s = 0.0f;
#pragma unroll
    for (int d = 0; d < VQ_D; ++d) {
        const float v = emb[k * VQ_D + d];
        s += v * v;
        embT[k * VQ_D + d] = (_Float16)v; // [k][d] layout for B-operand loads
    }
    esq[k] = s;
}

// ---------------------------------------------------------------------------
// Main: per-wave 16-row tile.  Scores via v_wmma_f32_16x16x32_f16,
// branchless u64-key argmin, f32 gather + loss partial.
// ---------------------------------------------------------------------------
__global__ __launch_bounds__(256) void vq_main(const float* __restrict__ w,
                                               const float* __restrict__ emb,
                                               const _Float16* __restrict__ embT,
                                               const float* __restrict__ esq,
                                               float* __restrict__ out,
                                               float* __restrict__ partial) {
    const int wave    = (int)((blockIdx.x * blockDim.x + threadIdx.x) >> 5);
    const int lane    = (int)(threadIdx.x & 31);
    const int half    = lane >> 4;        // 0: lanes 0-15, 1: lanes 16-31
    const int sub     = lane & 15;
    const int rowBase = wave * 16;

    // ---- A operand: 16x32 f16 slabs (slab 0 = d 0..31, slab 1 = d 32..63).
    // A k order used here (contiguous, same as B): lane(=row) 0-15 holds K 0..15 in elements
    // 0..15; lanes 16-31 hold K 16..31 for the same rows.
    const int    arow = rowBase + sub;
    const float* wrow = w + (size_t)arow * VQ_D + (half ? 16 : 0);
    v16h a[2];
#pragma unroll
    for (int s = 0; s < 2; ++s) {
#pragma unroll
        for (int c = 0; c < 2; ++c) {     // chunk base: d = s*32 + c*8 (+16 if half)
            const float* p = wrow + s * 32 + c * 8;
            const float4 f0 = *(const float4*)(p);
            const float4 f1 = *(const float4*)(p + 4);
            const int e = c * 8;
            a[s][e + 0] = (_Float16)f0.x;  a[s][e + 1] = (_Float16)f0.y;
            a[s][e + 2] = (_Float16)f0.z;  a[s][e + 3] = (_Float16)f0.w;
            a[s][e + 4] = (_Float16)f1.x;  a[s][e + 5] = (_Float16)f1.y;
            a[s][e + 6] = (_Float16)f1.z;  a[s][e + 7] = (_Float16)f1.w;
        }
    }

    float bestv[8];
    int   besti[8];
#pragma unroll
    for (int j = 0; j < 8; ++j) { bestv[j] = 3.4e38f; besti[j] = 0x7fffffff; }

    // B layout: lane%16 = code column, packed 16 halves = d 32*slab + 16*half
    // + 0..15 -> contiguous load from code-major embT.
    const _Float16* b0p = embT + (size_t)sub * VQ_D + half * 16;
    const _Float16* b1p = b0p + 32;

#pragma unroll 2
    for (int t = 0; t < 16; ++t) {        // 16 column tiles of 16 codes
        const v16h b0 = *(const v16h*)(b0p + t * 16 * VQ_D);
        const v16h b1 = *(const v16h*)(b1p + t * 16 * VQ_D);
        v8f acc = {};
        acc = __builtin_amdgcn_wmma_f32_16x16x32_f16(false, a[0], false, b0,
                                                     (short)0, acc, false, false);
        acc = __builtin_amdgcn_wmma_f32_16x16x32_f16(false, a[1], false, b1,
                                                     (short)0, acc, false, false);
        const int   col = t * 16 + sub;   // C layout: N = lane within half
        const float eq  = esq[col];
#pragma unroll
        for (int j = 0; j < 8; ++j) {     // C layout: VGPR j -> rows j / j+8
            const float score = fmaf(-2.0f, acc[j], eq);
            if (score < bestv[j]) { bestv[j] = score; besti[j] = col; }
        }
    }

    // Pack (score, col) into a monotonic 64-bit key: IEEE float -> ordered
    // uint in the high word, column index in the low word.  Lexicographic
    // u64 min == (min score, ties -> lowest index).  Fully branchless.
    unsigned long long key[8];
#pragma unroll
    for (int j = 0; j < 8; ++j) {
        unsigned u = __float_as_uint(bestv[j]);
        u ^= ((unsigned)((int)u >> 31)) | 0x80000000u;
        key[j] = ((unsigned long long)u << 32) | (unsigned)besti[j];
    }
    // Reduce across the 16 lanes of each half-wave (rows differ between halves).
#pragma unroll
    for (int j = 0; j < 8; ++j) {
#pragma unroll
        for (int m = 1; m < 16; m <<= 1) {
            const unsigned long long o =
                (unsigned long long)__shfl_xor((long long)key[j], m, 32);
            key[j] = (o < key[j]) ? o : key[j];
        }
    }

    // Gather codebook rows in f32, store output (non-temporal: write-once
    // stream, keep L2 for the weight re-read), accumulate loss partial.
    float lsum = 0.0f;
#pragma unroll
    for (int j = 0; j < 8; ++j) {
        const int orow = rowBase + half * 8 + j;
        const int idx  = (int)(unsigned)(key[j] & 0xffffffffull);
        const fvec4 q  = *(const fvec4*)(emb + (size_t)idx  * VQ_D + sub * 4);
        const fvec4 wv = *(const fvec4*)(w   + (size_t)orow * VQ_D + sub * 4);
        __builtin_nontemporal_store(q, (fvec4*)(out + (size_t)orow * VQ_D + sub * 4));
        const fvec4 d = q - wv;
        lsum += d.x * d.x + d.y * d.y + d.z * d.z + d.w * d.w;
    }
#pragma unroll
    for (int m = 1; m < 32; m <<= 1) lsum += __shfl_xor(lsum, m, 32);
    if (lane == 0) partial[wave] = lsum;
}

// ---------------------------------------------------------------------------
// Deterministic fixed-order reduction of 32768 partials -> scalar loss.
// ---------------------------------------------------------------------------
__global__ __launch_bounds__(256) void vq_loss_reduce(const float* __restrict__ partial,
                                                      float* __restrict__ loss_out) {
    __shared__ float sm[256];
    const int tid = threadIdx.x;
    float s = 0.0f;
    for (int k = 0; k < VQ_WAVES / 256; ++k) s += partial[tid * (VQ_WAVES / 256) + k];
    sm[tid] = s;
    __syncthreads();
    for (int off = 128; off > 0; off >>= 1) {
        if (tid < off) sm[tid] += sm[tid + off];
        __syncthreads();
    }
    if (tid == 0)
        *loss_out = sm[0] * (1.25f / ((float)VQ_N * (float)VQ_D));
}

extern "C" void kernel_launch(void* const* d_in, const int* in_sizes, int n_in,
                              void* d_out, int out_size, void* d_ws, size_t ws_size,
                              hipStream_t stream) {
    const float* w   = (const float*)d_in[0];   // [N, 64] f32
    const float* emb = (const float*)d_in[1];   // [256, 64] f32
    float* out = (float*)d_out;                 // [N*64] quantized + [1] vq_loss

    _Float16* embT = (_Float16*)d_ws;
    float*    esq  = (float*)((char*)d_ws + 32768);
    float*    part = (float*)((char*)d_ws + 33792);

    vq_prep<<<1, 256, 0, stream>>>(emb, embT, esq);
    vq_main<<<VQ_WAVES / 8, 256, 0, stream>>>(w, emb, embT, esq, out, part);
    vq_loss_reduce<<<1, 256, 0, stream>>>(part, out + (size_t)VQ_N * VQ_D);
}
